// DeepRecurrentModel_37263136260410
// MI455X (gfx1250) — matrix operand order repaired and checked by reading, not compiled, not run on hardware
//
#include <hip/hip_runtime.h>

typedef __attribute__((ext_vector_type(16))) __bf16 v16bf;
typedef __attribute__((ext_vector_type(8)))  float  v8f;

#define B_   64
#define T_   512
#define IN_  128
#define H_   1024
#define OUT_ 128

#if defined(__has_builtin)
#if __has_builtin(__builtin_amdgcn_sched_group_barrier)
#define SGB(mask, cnt, id) __builtin_amdgcn_sched_group_barrier(mask, cnt, id)
#endif
#endif
#ifndef SGB
#define SGB(mask, cnt, id)
#endif

// ---------------------------------------------------------------------------
// WMMA fragment helpers (layouts per CDNA5 ISA 7.12.2, wave32)
// ---------------------------------------------------------------------------

// A: 16x32 bf16, row-major source (lda elements). lane = row (l&15),
// halves at K offsets (l>>4)*8 and 16+(l>>4)*8 -> two 16B loads.
static __device__ __forceinline__ v16bf load_frag_a(const __bf16* __restrict__ A,
                                                    int lda, int row0, int k0) {
  int lane = threadIdx.x & 31;
  int r  = row0 + (lane & 15);
  int kb = k0 + ((lane >> 4) << 3);
  const __bf16* p = A + (size_t)r * lda + kb;
  union { uint4 q[2]; v16bf v; } u;
  u.q[0] = *reinterpret_cast<const uint4*>(p);
  u.q[1] = *reinterpret_cast<const uint4*>(p + 16);
  return u.v;
}

// B: 32x16 bf16; weights stored (N,K) row-major so each lane's 16 K-values
// for its column are contiguous. lane = col (l&15), halves at K offsets
// (l>>4)*8 and 16+(l>>4)*8 (same order as A).
static __device__ __forceinline__ v16bf load_frag_b(const __bf16* __restrict__ W,
                                                    int ldw, int col0, int k0) {
  int lane = threadIdx.x & 31;
  int c  = col0 + (lane & 15);
  int kb = k0 + ((lane >> 4) << 3);
  const __bf16* p = W + (size_t)c * ldw + kb;
  union { uint4 q[2]; v16bf v; } u;
  u.q[0] = *reinterpret_cast<const uint4*>(p);
  u.q[1] = *reinterpret_cast<const uint4*>(p + 16);
  return u.v;
}

static __device__ __forceinline__ v8f wmma_bf16(v16bf a, v16bf b, v8f c) {
  return __builtin_amdgcn_wmma_f32_16x16x32_bf16(false, a, false, b,
                                                 (short)0, c, false, false);
}

// ---------------------------------------------------------------------------
// Generic GEMM: C[M,N] = act(A[M,K](bf16) @ W[N,K](bf16)^T + bias)
// grid = (ceil(N/256), M/64), block = 256 (8 waves).
// Each wave: 32x64 tile (2 M-frags x 4 N-frags). K loop double-buffered and
// the scheduler is steered with sched_group_barrier so that VMEM reads for
// one half-step interleave with WMMAs of the other (partial s_wait_loadcnt).
// ---------------------------------------------------------------------------
template <bool RELU, bool OUT_BF16>
__global__ __launch_bounds__(256) void gemm_kernel(
    const __bf16* __restrict__ A, int lda,
    const __bf16* __restrict__ W, int ldw,
    const float* __restrict__ bias,
    float* __restrict__ Cf, __bf16* __restrict__ Cb, int ldc,
    int N, int K) {
  int wave = threadIdx.x >> 5;
  int lane = threadIdx.x & 31;
  int mwave = wave >> 2;          // 0..1
  int nwave = wave & 3;           // 0..3
  int wm = blockIdx.y * 64 + mwave * 32;
  int wn = blockIdx.x * 256 + nwave * 64;
  if (wn >= N) return;            // no LDS / barriers -> safe early exit

  v8f zero = {};
  v8f acc[2][4];
#pragma unroll
  for (int mt = 0; mt < 2; ++mt)
#pragma unroll
    for (int nt = 0; nt < 4; ++nt) acc[mt][nt] = zero;

  v16bf a0[2], b0[4];             // buffer 0
  v16bf a1[2], b1[4];             // buffer 1

  // prologue: fill buffer 0 with k = 0
  a0[0] = load_frag_a(A, lda, wm,      0);
  a0[1] = load_frag_a(A, lda, wm + 16, 0);
#pragma unroll
  for (int nt = 0; nt < 4; ++nt) b0[nt] = load_frag_b(W, ldw, wn + nt * 16, 0);

  for (int k = 0; k < K; k += 64) {
    // branchless refill offset for buffer 0 (keeps the body a single
    // scheduling region; last iteration re-reads k=0, result unused)
    int kn = (k + 64 < K) ? (k + 64) : 0;

    // loads for buffer 1 (k+32) -- K % 64 == 0 so always in range
    a1[0] = load_frag_a(A, lda, wm,      k + 32);
    a1[1] = load_frag_a(A, lda, wm + 16, k + 32);
#pragma unroll
    for (int nt = 0; nt < 4; ++nt) b1[nt] = load_frag_b(W, ldw, wn + nt * 16, k + 32);

    // WMMA on buffer 0 (k)
#pragma unroll
    for (int nt = 0; nt < 4; ++nt) {
      acc[0][nt] = wmma_bf16(a0[0], b0[nt], acc[0][nt]);
      acc[1][nt] = wmma_bf16(a0[1], b0[nt], acc[1][nt]);
    }

    // loads for buffer 0 (k+64)
    a0[0] = load_frag_a(A, lda, wm,      kn);
    a0[1] = load_frag_a(A, lda, wm + 16, kn);
#pragma unroll
    for (int nt = 0; nt < 4; ++nt) b0[nt] = load_frag_b(W, ldw, wn + nt * 16, kn);

    // WMMA on buffer 1 (k+32)
#pragma unroll
    for (int nt = 0; nt < 4; ++nt) {
      acc[0][nt] = wmma_bf16(a1[0], b1[nt], acc[0][nt]);
      acc[1][nt] = wmma_bf16(a1[1], b1[nt], acc[1][nt]);
    }

    // Pipeline shape: 24 VMEM reads + 16 WMMA per iteration, interleaved
    // as 8 x (3 loads, 2 wmma) so loads stay in flight across matrix ops.
#pragma unroll
    for (int g = 0; g < 8; ++g) {
      SGB(0x020, 3, 0);   // 3 VMEM reads
      SGB(0x008, 2, 0);   // 2 WMMA
    }
  }

  // Epilogue: C layout -> row = (l>>4)*8 + i, col = l&15
  int cbase = lane & 15;
  int rsub  = (lane >> 4) << 3;
#pragma unroll
  for (int nt = 0; nt < 4; ++nt) {
    int col = wn + nt * 16 + cbase;
    float bv = bias ? bias[col] : 0.0f;
#pragma unroll
    for (int mt = 0; mt < 2; ++mt) {
      int row0 = wm + mt * 16 + rsub;
#pragma unroll
      for (int i = 0; i < 8; ++i) {
        float v = acc[mt][nt][i] + bv;
        if (RELU) v = v > 0.0f ? v : 0.0f;
        size_t idx = (size_t)(row0 + i) * ldc + col;
        if (OUT_BF16) Cb[idx] = (__bf16)v;
        else          Cf[idx] = v;
      }
    }
  }
}

// ---------------------------------------------------------------------------
// Fused GRU step: gh = h @ Whh^T (+bhh) for the 3 gate slices, then gate math.
// grid = H/32 = 32 blocks, block = 256 (8 waves).
// wave: m = w>>1 -> batch rows m*16..+15 ; nc = w&1 -> 16 hidden cols.
// K loop double-buffered + sched_group_barrier interleave (K = 1024).
// ---------------------------------------------------------------------------
__global__ __launch_bounds__(256) void gru_step_kernel(
    const __bf16* __restrict__ hbf,       // (64,1024) bf16
    const __bf16* __restrict__ Whh,       // (3072,1024) bf16 (N,K)
    const float* __restrict__ bhh,        // (3072)
    const float* __restrict__ gi,         // (B,T,3H) fp32, includes bih
    int t,
    const float* __restrict__ hprev,      // (64,1024) fp32
    float* __restrict__ hnew,             // (64,1024) fp32
    __bf16* __restrict__ hnewbf) {        // (64,1024) bf16
  int wave = threadIdx.x >> 5;
  int lane = threadIdx.x & 31;
  int m  = wave >> 1;                     // 0..3
  int nc = wave & 1;                      // 0..1
  int j0 = blockIdx.x * 32 + nc * 16;     // hidden column base
  int colR = j0, colZ = j0 + H_, colN = j0 + 2 * H_;

  v8f ar = {}, az = {}, an = {};
  v16bf a0, br0, bz0, bn0;
  v16bf a1, br1, bz1, bn1;

  a0  = load_frag_a(hbf, H_, m * 16, 0);
  br0 = load_frag_b(Whh, H_, colR, 0);
  bz0 = load_frag_b(Whh, H_, colZ, 0);
  bn0 = load_frag_b(Whh, H_, colN, 0);

  for (int k = 0; k < H_; k += 64) {
    int kn = (k + 64 < H_) ? (k + 64) : 0;

    a1  = load_frag_a(hbf, H_, m * 16, k + 32);
    br1 = load_frag_b(Whh, H_, colR, k + 32);
    bz1 = load_frag_b(Whh, H_, colZ, k + 32);
    bn1 = load_frag_b(Whh, H_, colN, k + 32);

    ar = wmma_bf16(a0, br0, ar);
    az = wmma_bf16(a0, bz0, az);
    an = wmma_bf16(a0, bn0, an);

    a0  = load_frag_a(hbf, H_, m * 16, kn);
    br0 = load_frag_b(Whh, H_, colR, kn);
    bz0 = load_frag_b(Whh, H_, colZ, kn);
    bn0 = load_frag_b(Whh, H_, colN, kn);

    ar = wmma_bf16(a1, br1, ar);
    az = wmma_bf16(a1, bz1, az);
    an = wmma_bf16(a1, bn1, an);

    // 16 VMEM reads + 6 WMMA per iteration
#pragma unroll
    for (int g = 0; g < 2; ++g) {
      SGB(0x020, 3, 0); SGB(0x008, 1, 0);
      SGB(0x020, 3, 0); SGB(0x008, 1, 0);
      SGB(0x020, 2, 0); SGB(0x008, 1, 0);
    }
  }

  int c    = lane & 15;
  int rsub = (lane >> 4) << 3;
  int jcol = j0 + c;
  float bR = bhh[colR + c], bZ = bhh[colZ + c], bN = bhh[colN + c];
#pragma unroll
  for (int i = 0; i < 8; ++i) {
    int b = m * 16 + rsub + i;            // batch index
    size_t gbase = ((size_t)b * T_ + t) * (3 * H_);
    float gr = gi[gbase + colR + c];
    float gz = gi[gbase + colZ + c];
    float gn = gi[gbase + colN + c];
    float R = 1.0f / (1.0f + __expf(-(gr + ar[i] + bR)));
    float Z = 1.0f / (1.0f + __expf(-(gz + az[i] + bZ)));
    float pre = gn + R * (an[i] + bN);
    float ex = __expf(2.0f * pre);
    float Nn = (ex - 1.0f) / (ex + 1.0f); // tanh
    float hp = hprev[(size_t)b * H_ + jcol];
    float hv = (1.0f - Z) * Nn + Z * hp;
    hnew[(size_t)b * H_ + jcol]   = hv;
    hnewbf[(size_t)b * H_ + jcol] = (__bf16)hv;
  }
}

// ---------------------------------------------------------------------------
// Small prep / utility kernels
// ---------------------------------------------------------------------------
// out (N,K) bf16 <- in (K,N) fp32 (transpose + convert)
__global__ void cvt_transpose_kernel(const float* __restrict__ in,
                                     __bf16* __restrict__ out, int K, int N) {
  int idx = blockIdx.x * blockDim.x + threadIdx.x;
  if (idx >= N * K) return;
  int n = idx / K, k = idx - n * K;
  out[idx] = (__bf16)in[(size_t)k * N + n];
}

__global__ void cvt_kernel(const float* __restrict__ in,
                           __bf16* __restrict__ out, int n) {
  int idx = blockIdx.x * blockDim.x + threadIdx.x;
  if (idx < n) out[idx] = (__bf16)in[idx];
}

__global__ void normalize_kernel(const float* __restrict__ x,
                                 const float* __restrict__ mean,
                                 const float* __restrict__ stdv,
                                 __bf16* __restrict__ out, int total) {
  int idx = blockIdx.x * blockDim.x + threadIdx.x;
  if (idx >= total) return;
  int c = idx & (IN_ - 1);
  out[idx] = (__bf16)((x[idx] - mean[c]) / stdv[c]);
}

__global__ void init_h_kernel(const float* __restrict__ carry,
                              float* __restrict__ hf,
                              __bf16* __restrict__ hbf, int n) {
  int idx = blockIdx.x * blockDim.x + threadIdx.x;
  if (idx >= n) return;
  float v = carry[idx];
  hf[idx]  = v;
  hbf[idx] = (__bf16)v;
}

__global__ void copy_kernel(const float* __restrict__ src,
                            float* __restrict__ dst, int n) {
  int idx = blockIdx.x * blockDim.x + threadIdx.x;
  if (idx < n) dst[idx] = src[idx];
}

// ---------------------------------------------------------------------------
// Host orchestration
// ---------------------------------------------------------------------------
extern "C" void kernel_launch(void* const* d_in, const int* in_sizes, int n_in,
                              void* d_out, int out_size, void* d_ws, size_t ws_size,
                              hipStream_t stream) {
  const float* x     = (const float*)d_in[0];
  const float* carry = (const float*)d_in[1];
  const float* mean  = (const float*)d_in[2];
  const float* stdv  = (const float*)d_in[3];
  const float* We0   = (const float*)d_in[4];
  const float* be0   = (const float*)d_in[5];
  const float* We1   = (const float*)d_in[6];
  const float* be1   = (const float*)d_in[7];
  const float* We2   = (const float*)d_in[8];
  const float* be2   = (const float*)d_in[9];
  const float* Wih   = (const float*)d_in[10];
  const float* bih   = (const float*)d_in[11];
  const float* Whh   = (const float*)d_in[12];
  const float* bhh   = (const float*)d_in[13];
  const float* Wd0   = (const float*)d_in[14];
  const float* bd0   = (const float*)d_in[15];
  const float* Wd1   = (const float*)d_in[16];
  const float* bd1   = (const float*)d_in[17];
  const float* Wd2   = (const float*)d_in[18];
  const float* bd2   = (const float*)d_in[19];
  float* out = (float*)d_out;

  const int BT = B_ * T_;               // 32768
  char* ws = (char*)d_ws;
  size_t off = 0;
  auto alloc = [&](size_t bytes) -> void* {
    void* p = ws + off;
    off += (bytes + 255) & ~(size_t)255;
    return p;
  };

  __bf16* xn    = (__bf16*)alloc((size_t)BT * IN_ * 2);
  __bf16* ebuf0 = (__bf16*)alloc((size_t)BT * H_ * 2);
  __bf16* ebuf1 = (__bf16*)alloc((size_t)BT * H_ * 2);
  float*  gi    = (float*)alloc((size_t)BT * 3 * H_ * 4);
  __bf16* We0t  = (__bf16*)alloc((size_t)H_ * IN_ * 2);
  __bf16* We1t  = (__bf16*)alloc((size_t)H_ * H_ * 2);
  __bf16* We2t  = (__bf16*)alloc((size_t)H_ * H_ * 2);
  __bf16* Wd0t  = (__bf16*)alloc((size_t)H_ * H_ * 2);
  __bf16* Wd1t  = (__bf16*)alloc((size_t)H_ * H_ * 2);
  __bf16* Wd2t  = (__bf16*)alloc((size_t)OUT_ * H_ * 2);
  __bf16* Wihb  = (__bf16*)alloc((size_t)3 * H_ * H_ * 2);
  __bf16* Whhb  = (__bf16*)alloc((size_t)3 * H_ * H_ * 2);
  float*  hf0   = (float*)alloc((size_t)B_ * H_ * 4);
  float*  hf1   = (float*)alloc((size_t)B_ * H_ * 4);
  __bf16* hb0   = (__bf16*)alloc((size_t)B_ * H_ * 2);
  __bf16* hb1   = (__bf16*)alloc((size_t)B_ * H_ * 2);
  __bf16* d0b   = (__bf16*)alloc((size_t)B_ * H_ * 2);
  __bf16* d1b   = (__bf16*)alloc((size_t)B_ * H_ * 2);

  // ---- weight prep (bf16, transposed to (N,K) where needed) ----
  cvt_transpose_kernel<<<(H_*IN_ + 255)/256, 256, 0, stream>>>(We0, We0t, IN_, H_);
  cvt_transpose_kernel<<<(H_*H_  + 255)/256, 256, 0, stream>>>(We1, We1t, H_, H_);
  cvt_transpose_kernel<<<(H_*H_  + 255)/256, 256, 0, stream>>>(We2, We2t, H_, H_);
  cvt_transpose_kernel<<<(H_*H_  + 255)/256, 256, 0, stream>>>(Wd0, Wd0t, H_, H_);
  cvt_transpose_kernel<<<(H_*H_  + 255)/256, 256, 0, stream>>>(Wd1, Wd1t, H_, H_);
  cvt_transpose_kernel<<<(OUT_*H_+ 255)/256, 256, 0, stream>>>(Wd2, Wd2t, H_, OUT_);
  cvt_kernel<<<(3*H_*H_ + 255)/256, 256, 0, stream>>>(Wih, Wihb, 3*H_*H_);
  cvt_kernel<<<(3*H_*H_ + 255)/256, 256, 0, stream>>>(Whh, Whhb, 3*H_*H_);

  // ---- normalize + encoder ----
  normalize_kernel<<<(BT*IN_ + 255)/256, 256, 0, stream>>>(x, mean, stdv, xn, BT*IN_);

  dim3 gE(H_ / 256, BT / 64);   // (4, 512)
  gemm_kernel<true,  true><<<gE, 256, 0, stream>>>(xn,    IN_, We0t, IN_, be0, nullptr, ebuf0, H_, H_, IN_);
  gemm_kernel<true,  true><<<gE, 256, 0, stream>>>(ebuf0, H_,  We1t, H_,  be1, nullptr, ebuf1, H_, H_, H_);
  gemm_kernel<false, true><<<gE, 256, 0, stream>>>(ebuf1, H_,  We2t, H_,  be2, nullptr, ebuf0, H_, H_, H_);

  // ---- hoisted input-gate GEMM: gi = e2 @ Wih^T + bih  (fp32 out) ----
  dim3 gGI(3 * H_ / 256, BT / 64);  // (12, 512)
  gemm_kernel<false, false><<<gGI, 256, 0, stream>>>(ebuf0, H_, Wihb, H_, bih, gi, nullptr, 3*H_, 3*H_, H_);

  // ---- GRU scan: 512 fused step launches (stream order = grid sync) ----
  init_h_kernel<<<(B_*H_ + 255)/256, 256, 0, stream>>>(carry, hf0, hb0, B_*H_);
  float*  hf[2] = {hf0, hf1};
  __bf16* hb[2] = {hb0, hb1};
  int cur = 0;
  for (int t = 0; t < T_; ++t) {
    gru_step_kernel<<<H_ / 32, 256, 0, stream>>>(hb[cur], Whhb, bhh, gi, t,
                                                 hf[cur], hf[1 - cur], hb[1 - cur]);
    cur ^= 1;
  }

  // ---- outputs: h_final into d_out[8192:], decoder into d_out[:8192] ----
  copy_kernel<<<(B_*H_ + 255)/256, 256, 0, stream>>>(hf[cur], out + B_ * OUT_, B_ * H_);

  dim3 gD(H_ / 256, 1);   // (4,1)
  gemm_kernel<true,  true><<<gD, 256, 0, stream>>>(hb[cur], H_, Wd0t, H_, bd0, nullptr, d0b, H_, H_, H_);
  gemm_kernel<true,  true><<<gD, 256, 0, stream>>>(d0b,     H_, Wd1t, H_, bd1, nullptr, d1b, H_, H_, H_);
  dim3 gO(1, 1);
  gemm_kernel<false, false><<<gO, 256, 0, stream>>>(d1b, H_, Wd2t, H_, bd2, out, nullptr, OUT_, OUT_, H_);
}
